// InterSentenceGNN_58884001628476
// MI455X (gfx1250) — matrix operand order repaired and checked by reading, not compiled, not run on hardware
//
#include <hip/hip_runtime.h>
#include <hip/hip_bf16.h>

typedef __attribute__((ext_vector_type(16))) _Float16 v16h;
typedef __attribute__((ext_vector_type(8)))  _Float16 v8h;
typedef __attribute__((ext_vector_type(8)))  float    v8f;

#define NODE_DIM 768
#define HID1 512
#define OUTD 256

union F16x16 { v16h v; v8h h2[2]; _Float16 e[16]; };

// ---------------------------------------------------------------------------
// float atomic max via integer ordering trick (bit-pattern monotone for IEEE)
// ---------------------------------------------------------------------------
__device__ __forceinline__ void atomicMaxF(float* addr, float val) {
  if (val >= 0.f) {
    atomicMax((int*)addr, __float_as_int(val));
  } else {
    atomicMin((unsigned int*)addr, __float_as_uint(val));
  }
}

__global__ void fill_f32_kernel(float* __restrict__ dst, float v, int n) {
  int i = blockIdx.x * blockDim.x + threadIdx.x;
  if (i < n) dst[i] = v;
}

// ---------------------------------------------------------------------------
// Pack fp32 row-major W[K x Nc] into fp16 WMMA B-fragment layout.
// Fragment (ct, kt) covers cols [ct*16, ct*16+16), Ks [kt*32, kt*32+32).
// Lane L of a fragment holds column n = ct*16 + (L&15),
// K values kb..kb+15 with kb = kt*32 + (L>>4)*16, stored contiguously.
// ---------------------------------------------------------------------------
__global__ void pack_B_kernel(const float* __restrict__ W, _Float16* __restrict__ P,
                              int K, int Nc) {
  const int idx = blockIdx.x * blockDim.x + threadIdx.x;
  const int KT = K >> 5;
  const int NT = Nc >> 4;
  if (idx >= NT * KT * 32) return;
  const int lane = idx & 31;
  const int frag = idx >> 5;
  const int kt = frag % KT;
  const int ct = frag / KT;
  const int n  = (ct << 4) + (lane & 15);
  const int kb = (kt << 5) + ((lane >> 4) << 4);
  _Float16* dst = P + ((size_t)frag * 32 + lane) * 16;
  #pragma unroll
  for (int j = 0; j < 16; ++j)
    dst[j] = (_Float16)W[(size_t)(kb + j) * Nc + n];
}

// ---------------------------------------------------------------------------
// node gate: w = sigmoid(relu(rel@w1+b1)@w2+b2); h = (x*w) as fp16
// ---------------------------------------------------------------------------
__global__ __launch_bounds__(64)
void gate_kernel(const float* __restrict__ x, const float* __restrict__ rel,
                 const float* __restrict__ w1, const float* __restrict__ b1,
                 const float* __restrict__ w2, const float* __restrict__ b2,
                 _Float16* __restrict__ hbf) {
  const int node = blockIdx.x;
  const int t = threadIdx.x;            // 0..63
  __shared__ float red[64];
  const float r0 = rel[node * 3 + 0];
  const float r1 = rel[node * 3 + 1];
  const float r2 = rel[node * 3 + 2];
  float hj = r0 * w1[t] + r1 * w1[64 + t] + r2 * w1[128 + t] + b1[t];
  hj = fmaxf(hj, 0.f);
  red[t] = hj * w2[t];
  __syncthreads();
  for (int s = 32; s > 0; s >>= 1) {
    if (t < s) red[t] += red[t + s];
    __syncthreads();
  }
  const float z = red[0] + b2[0];
  const float w = 1.f / (1.f + __expf(-z));
  for (int c = t; c < NODE_DIM; c += 64)
    hbf[(size_t)node * NODE_DIM + c] = (_Float16)(x[(size_t)node * NODE_DIM + c] * w);
}

// ---------------------------------------------------------------------------
// WMMA GEMM: C[M x Nc] = A[M x K] @ W[K x Nc] + bias
// A fp16 row-major; W pre-packed in fragment layout (pack_B_kernel).
// One wave computes a 16x64 strip (4 accumulators, A fragment reused 4x).
// Double-buffered: fragments for step kt+1 are prefetched into a second
// register buffer before the 4 WMMAs of step kt, so the loads overlap the
// matrix pipe instead of forcing s_wait_loadcnt 0 per WMMA.
// KT must be even (768/32=24, 512/32=16 both are).
// ---------------------------------------------------------------------------
__global__ __launch_bounds__(128)
void gemm_wmma_f16(const _Float16* __restrict__ A, const _Float16* __restrict__ P,
                   const float* __restrict__ bias, float* __restrict__ C,
                   int M, int K, int Nc) {
  const int lane = threadIdx.x & 31;
  const int wave = blockIdx.x * (blockDim.x >> 5) + (threadIdx.x >> 5);
  const int ngroups = Nc >> 6;          // 64-wide column groups
  const int mt = wave / ngroups;
  const int ng = wave % ngroups;
  if (mt * 16 >= M) return;
  const int KT   = K >> 5;
  const int lrow = lane & 15;
  const int lhi  = lane >> 4;           // 0 or 1

  // A: lane<16 -> K [0..15]; lane>=16 -> K [16..31]
  const _Float16* arow = A + (size_t)(mt * 16 + lrow) * K + lhi * 16;
  // B fragments for this wave's 4 col tiles: ((ct*KT + kt)*32 + lane)*16
  const _Float16* bbase = P + (((size_t)(ng << 2) * KT) * 32 + lane) * 16;

  v8f acc[4] = {v8f{}, v8f{}, v8f{}, v8f{}};
  F16x16 a[2];
  F16x16 b[2][4];

  auto loadA = [&](int buf, int kt) {
    const v8h* ap = reinterpret_cast<const v8h*>(arow + (kt << 5));
    a[buf].h2[0] = ap[0];
    a[buf].h2[1] = ap[1];               // +8 halfs
  };
  auto loadB = [&](int buf, int kt) {
    #pragma unroll
    for (int t = 0; t < 4; ++t) {
      const v8h* bp = reinterpret_cast<const v8h*>(
          bbase + ((size_t)t * KT + kt) * 512);   // 32 lanes * 16 halfs
      b[buf][t].h2[0] = bp[0];
      b[buf][t].h2[1] = bp[1];
    }
  };
  auto mm = [&](int buf) {
    #pragma unroll
    for (int t = 0; t < 4; ++t)
      acc[t] = __builtin_amdgcn_wmma_f32_16x16x32_f16(
          false, a[buf].v, false, b[buf][t].v, (short)0, acc[t], false, false);
  };

  loadA(0, 0);
  loadB(0, 0);
  for (int kt = 0; kt < KT; kt += 2) {
    loadA(1, kt + 1);
    loadB(1, kt + 1);
    mm(0);
    if (kt + 2 < KT) {
      loadA(0, kt + 2);
      loadB(0, kt + 2);
    }
    mm(1);
  }

  #pragma unroll
  for (int t = 0; t < 4; ++t) {
    const int ncol = (ng << 6) + (t << 4) + lrow;
    const float bs = bias[ncol];
    #pragma unroll
    for (int v = 0; v < 8; ++v) {
      const int mrow = mt * 16 + lhi * 8 + v;   // C/D layout: M = 8*lhi + vgpr
      C[(size_t)mrow * Nc + ncol] = acc[t][v] + bs;
    }
  }
}

// ---------------------------------------------------------------------------
// edge pass 1: logits + segment max (wave per edge; self-loops appended)
// channels interleaved: lane reads c = lane + 32*j  -> fully coalesced
// ---------------------------------------------------------------------------
template <int HIDv, int HEADS>
__global__ __launch_bounds__(128)
void edge_logits_t(const float* __restrict__ xl, const float* __restrict__ xr,
                   const float* __restrict__ att, const int* __restrict__ ei,
                   int E, int Ntot, float* __restrict__ logits,
                   float* __restrict__ maxb) {
  constexpr int J   = HIDv / 32;
  constexpr int JPH = (HIDv / HEADS) / 32;   // j-steps per head
  const int lane = threadIdx.x & 31;
  const int e = blockIdx.x * (blockDim.x >> 5) + (threadIdx.x >> 5);
  const int EE = E + Ntot;
  if (e >= EE) return;
  const int s = (e < E) ? ei[e]     : (e - E);
  const int d = (e < E) ? ei[E + e] : (e - E);

  float part[HEADS];
  #pragma unroll
  for (int h = 0; h < HEADS; ++h) part[h] = 0.f;
  #pragma unroll
  for (int j = 0; j < J; ++j) {
    const int c = lane + 32 * j;
    float v = xl[(size_t)s * HIDv + c] + xr[(size_t)d * HIDv + c];
    v = (v > 0.f) ? v : 0.2f * v;       // leaky_relu(0.2)
    part[j / JPH] += v * att[c];
  }
  #pragma unroll
  for (int h = 0; h < HEADS; ++h)
    for (int off = 1; off < 32; off <<= 1)
      part[h] += __shfl_xor(part[h], off, 32);
  if (lane == 0) {
    #pragma unroll
    for (int h = 0; h < HEADS; ++h) {
      logits[(size_t)e * HEADS + h] = part[h];
      atomicMaxF(&maxb[(size_t)d * HEADS + h], part[h]);
    }
  }
}

// ---------------------------------------------------------------------------
// edge pass 2: exp(logit - max[dst]) + segment sum of denominators
// ---------------------------------------------------------------------------
__global__ void edge_exp_kernel(float* __restrict__ logits,
                                const float* __restrict__ maxb,
                                float* __restrict__ denom,
                                const int* __restrict__ ei,
                                int E, int Ntot, int HEADS) {
  const int idx = blockIdx.x * blockDim.x + threadIdx.x;
  const int EE = E + Ntot;
  if (idx >= EE * HEADS) return;
  const int e = idx / HEADS;
  const int h = idx - e * HEADS;
  const int d = (e < E) ? ei[E + e] : (e - E);
  const float a = __expf(logits[idx] - maxb[(size_t)d * HEADS + h]);
  logits[idx] = a;                      // overwrite with exp value
  atomicAdd(&denom[(size_t)d * HEADS + h], a);
}

// ---------------------------------------------------------------------------
// edge pass 3: out[dst] += alpha * xl[src]  (wave per edge, coalesced)
// ---------------------------------------------------------------------------
template <int HIDv, int HEADS>
__global__ __launch_bounds__(128)
void edge_agg_t(const float* __restrict__ xl, const float* __restrict__ expv,
                const float* __restrict__ denom, const int* __restrict__ ei,
                int E, int Ntot, float* __restrict__ out) {
  constexpr int J   = HIDv / 32;
  constexpr int JPH = (HIDv / HEADS) / 32;
  const int lane = threadIdx.x & 31;
  const int e = blockIdx.x * (blockDim.x >> 5) + (threadIdx.x >> 5);
  const int EE = E + Ntot;
  if (e >= EE) return;
  const int s = (e < E) ? ei[e]     : (e - E);
  const int d = (e < E) ? ei[E + e] : (e - E);

  float alpha[HEADS];
  #pragma unroll
  for (int h = 0; h < HEADS; ++h)
    alpha[h] = expv[(size_t)e * HEADS + h] / denom[(size_t)d * HEADS + h];
  #pragma unroll
  for (int j = 0; j < J; ++j) {
    const int c = lane + 32 * j;
    atomicAdd(&out[(size_t)d * HIDv + c],
              xl[(size_t)s * HIDv + c] * alpha[j / JPH]);
  }
}

// ---------------------------------------------------------------------------
// bias + ELU -> fp16 (layer-1 output feeding layer-2 GEMM)
// ---------------------------------------------------------------------------
__global__ void bias_elu_f16_kernel(const float* __restrict__ agg,
                                    const float* __restrict__ bias,
                                    _Float16* __restrict__ out,
                                    int total, int HIDv) {
  int idx = blockIdx.x * blockDim.x + threadIdx.x;
  if (idx >= total) return;
  const int c = idx % HIDv;
  float v = agg[idx] + bias[c];
  v = (v > 0.f) ? v : expm1f(v);
  out[idx] = (_Float16)v;
}

// ---------------------------------------------------------------------------
// LayerNorm on the 128 gathered rows only; write fp32 output
// ---------------------------------------------------------------------------
__global__ __launch_bounds__(256)
void finalize_kernel(const float* __restrict__ agg2, const float* __restrict__ bias2,
                     const float* __restrict__ ln_g, const float* __restrict__ ln_b,
                     const int* __restrict__ last, float* __restrict__ out) {
  const int b = blockIdx.x;
  const int t = threadIdx.x;            // 0..255
  const int row = last[b];
  const float v = agg2[(size_t)row * OUTD + t] + bias2[t];
  __shared__ float s[256];
  s[t] = v; __syncthreads();
  for (int st = 128; st > 0; st >>= 1) {
    if (t < st) s[t] += s[t + st];
    __syncthreads();
  }
  const float mu = s[0] * (1.f / OUTD);
  __syncthreads();
  const float d = v - mu;
  s[t] = d * d; __syncthreads();
  for (int st = 128; st > 0; st >>= 1) {
    if (t < st) s[t] += s[t + st];
    __syncthreads();
  }
  const float var = s[0] * (1.f / OUTD);
  out[(size_t)b * OUTD + t] = d * rsqrtf(var + 1e-5f) * ln_g[t] + ln_b[t];
}

// ---------------------------------------------------------------------------
// launch
// ---------------------------------------------------------------------------
extern "C" void kernel_launch(void* const* d_in, const int* in_sizes, int n_in,
                              void* d_out, int out_size, void* d_ws, size_t ws_size,
                              hipStream_t stream) {
  const float* x    = (const float*)d_in[0];
  const float* rel  = (const float*)d_in[1];
  const float* w1   = (const float*)d_in[2];
  const float* b1   = (const float*)d_in[3];
  const float* w2   = (const float*)d_in[4];
  const float* b2   = (const float*)d_in[5];
  const float* Wl1  = (const float*)d_in[6];
  const float* bl1  = (const float*)d_in[7];
  const float* Wr1  = (const float*)d_in[8];
  const float* br1  = (const float*)d_in[9];
  const float* att1 = (const float*)d_in[10];
  const float* bias1= (const float*)d_in[11];
  const float* Wl2  = (const float*)d_in[12];
  const float* bl2  = (const float*)d_in[13];
  const float* Wr2  = (const float*)d_in[14];
  const float* br2  = (const float*)d_in[15];
  const float* att2 = (const float*)d_in[16];
  const float* bias2= (const float*)d_in[17];
  const float* ln_g = (const float*)d_in[18];
  const float* ln_b = (const float*)d_in[19];
  const int*   ei   = (const int*)d_in[20];
  const int*   last = (const int*)d_in[21];
  float* out = (float*)d_out;

  const int N  = in_sizes[0] / NODE_DIM;     // 10240
  const int E  = in_sizes[20] / 2;           // unique edges (no self loops)
  const int EE = E + N;                      // with self loops
  const int Bn = in_sizes[21];               // 128

  // ---- workspace carve-out ----
  char* ws = (char*)d_ws;
  size_t off = 0;
  auto take = [&](size_t bytes) -> void* {
    void* p = ws + off;
    off = (off + bytes + 255) & ~(size_t)255;
    return p;
  };
  _Float16* hbf   = (_Float16*)take((size_t)N * NODE_DIM * 2);
  _Float16* Wl1p  = (_Float16*)take((size_t)NODE_DIM * HID1 * 2);
  _Float16* Wr1p  = (_Float16*)take((size_t)NODE_DIM * HID1 * 2);
  _Float16* Wl2p  = (_Float16*)take((size_t)HID1 * OUTD * 2);
  _Float16* Wr2p  = (_Float16*)take((size_t)HID1 * OUTD * 2);
  float* xl1   = (float*)take((size_t)N * HID1 * 4);
  float* xr1   = (float*)take((size_t)N * HID1 * 4);
  float* log1  = (float*)take((size_t)EE * 8 * 4);
  float* max1  = (float*)take((size_t)N * 8 * 4);
  float* den1  = (float*)take((size_t)N * 8 * 4);
  float* agg1  = (float*)take((size_t)N * HID1 * 4);
  _Float16* h1h = (_Float16*)take((size_t)N * HID1 * 2);
  float* xl2   = (float*)take((size_t)N * OUTD * 4);
  float* xr2   = (float*)take((size_t)N * OUTD * 4);
  float* log2b = (float*)take((size_t)EE * 4);
  float* max2  = (float*)take((size_t)N * 4);
  float* den2  = (float*)take((size_t)N * 4);
  float* agg2  = (float*)take((size_t)N * OUTD * 4);

  // ---- pack weights into WMMA B-fragment layout (fp32 -> fp16 fused) ----
  {
    int t1 = (HID1 / 16) * (NODE_DIM / 32) * 32;   // threads for 768x512
    int t2 = (OUTD / 16) * (HID1 / 32) * 32;       // threads for 512x256
    pack_B_kernel<<<(t1 + 255) / 256, 256, 0, stream>>>(Wl1, Wl1p, NODE_DIM, HID1);
    pack_B_kernel<<<(t1 + 255) / 256, 256, 0, stream>>>(Wr1, Wr1p, NODE_DIM, HID1);
    pack_B_kernel<<<(t2 + 255) / 256, 256, 0, stream>>>(Wl2, Wl2p, HID1, OUTD);
    pack_B_kernel<<<(t2 + 255) / 256, 256, 0, stream>>>(Wr2, Wr2p, HID1, OUTD);
  }

  // ---- node gate -> fp16 h ----
  gate_kernel<<<N, 64, 0, stream>>>(x, rel, w1, b1, w2, b2, hbf);

  // ---- layer 1 GEMMs (WMMA) ----
  {
    int waves = (N / 16) * (HID1 / 64);
    int blocks = (waves + 3) / 4;
    gemm_wmma_f16<<<blocks, 128, 0, stream>>>(hbf, Wl1p, bl1, xl1, N, NODE_DIM, HID1);
    gemm_wmma_f16<<<blocks, 128, 0, stream>>>(hbf, Wr1p, br1, xr1, N, NODE_DIM, HID1);
  }

  // ---- layer 1 segment softmax + aggregate ----
  hipMemsetAsync(den1, 0, (size_t)N * 8 * 4, stream);
  hipMemsetAsync(agg1, 0, (size_t)N * HID1 * 4, stream);
  fill_f32_kernel<<<(N * 8 + 255) / 256, 256, 0, stream>>>(max1, -1e30f, N * 8);
  {
    int eblocks = (EE + 3) / 4;
    edge_logits_t<HID1, 8><<<eblocks, 128, 0, stream>>>(xl1, xr1, att1, ei, E, N,
                                                        log1, max1);
    int tot = EE * 8;
    edge_exp_kernel<<<(tot + 255) / 256, 256, 0, stream>>>(log1, max1, den1, ei, E, N, 8);
    edge_agg_t<HID1, 8><<<eblocks, 128, 0, stream>>>(xl1, log1, den1, ei, E, N, agg1);
  }

  // ---- bias + ELU -> fp16 ----
  {
    int tot = N * HID1;
    bias_elu_f16_kernel<<<(tot + 255) / 256, 256, 0, stream>>>(agg1, bias1, h1h, tot, HID1);
  }

  // ---- layer 2 GEMMs (WMMA) ----
  {
    int waves = (N / 16) * (OUTD / 64);
    int blocks = (waves + 3) / 4;
    gemm_wmma_f16<<<blocks, 128, 0, stream>>>(h1h, Wl2p, bl2, xl2, N, HID1, OUTD);
    gemm_wmma_f16<<<blocks, 128, 0, stream>>>(h1h, Wr2p, br2, xr2, N, HID1, OUTD);
  }

  // ---- layer 2 segment softmax + aggregate (1 head, 256 ch) ----
  hipMemsetAsync(den2, 0, (size_t)N * 4, stream);
  hipMemsetAsync(agg2, 0, (size_t)N * OUTD * 4, stream);
  fill_f32_kernel<<<(N + 255) / 256, 256, 0, stream>>>(max2, -1e30f, N);
  {
    int eblocks = (EE + 3) / 4;
    edge_logits_t<OUTD, 1><<<eblocks, 128, 0, stream>>>(xl2, xr2, att2, ei, E, N,
                                                        log2b, max2);
    edge_exp_kernel<<<(EE + 255) / 256, 256, 0, stream>>>(log2b, max2, den2, ei, E, N, 1);
    edge_agg_t<OUTD, 1><<<eblocks, 128, 0, stream>>>(xl2, log2b, den2, ei, E, N, agg2);
  }

  // ---- LayerNorm + gather ----
  finalize_kernel<<<Bn, 256, 0, stream>>>(agg2, bias2, ln_g, ln_b, last, out);
}
